// MyModel_76862734729434
// MI455X (gfx1250) — matrix operand order repaired and checked by reading, not compiled, not run on hardware
//
#include <hip/hip_runtime.h>

// ---------------- problem constants ----------------
#define BB      8192
#define HID     4096
#define NH      32
#define NKV     8
#define HD      128
#define RBIT    128
#define QKVN    6144
#define NQK     5120            // q+k columns only (v never used)
#define NPAIR   (BB * NH)       // 262144
#define INV_SQRT_D 0.08838834764831845f

// ---------------- vector types ----------------
typedef __attribute__((ext_vector_type(4)))  __bf16 v4bf;
typedef __attribute__((ext_vector_type(8)))  __bf16 v8bf;
typedef __attribute__((ext_vector_type(16))) __bf16 v16bf;
typedef __attribute__((ext_vector_type(8)))  float  v8f;

// ---------------- workspace layout ----------------
#define WBT_BYTES ((size_t)NQK * HID * 2)          // 40 MB  bf16 wBt[n][k]
#define HWT_BYTES ((size_t)RBIT * HD * 2)          // 32 KB  bf16 hwT[j][d]
#define QK_OFF    (WBT_BYTES + HWT_BYTES)
// qk fp32: 8192*5120*4 = 160 MB  -> total ~200 MB

// ---------------- fragment loaders (WMMA bf16 per-lane layouts) ----------------
// A (16x32, 16-bit): lane L holds M=L%16; half=L/16; elems 0..7 -> K=half*8+0..7,
// elems 8..15 -> K=16+half*8+0..7
__device__ __forceinline__ v16bf load_frag_a(const __bf16* p, int row, int half, int stride) {
    const v8bf lo = *(const v8bf*)(p + row * stride + half * 8);
    const v8bf hi = *(const v8bf*)(p + row * stride + 16 + half * 8);
    return __builtin_shufflevector(lo, hi, 0,1,2,3,4,5,6,7,8,9,10,11,12,13,14,15);
}
// B (32x16, 16-bit): lane L holds N=L%16; half=L/16; elems 0..7 -> K=half*8+0..7,
// elems 8..15 -> K=16+half*8+0..7
// Stored transposed in LDS (row = N, contiguous K), so two 8-bf16 chunks.
__device__ __forceinline__ v16bf load_frag_b(const __bf16* p, int row, int half, int stride) {
    const v8bf lo = *(const v8bf*)(p + row * stride + half * 8);
    const v8bf hi = *(const v8bf*)(p + row * stride + 16 + half * 8);
    return __builtin_shufflevector(lo, hi, 0,1,2,3,4,5,6,7,8,9,10,11,12,13,14,15);
}

// ================= kernel 0a: proj_w[:, :5120] -> wBt[n][k] bf16 =================
__global__ __launch_bounds__(256) void k_transpose_projw(const float* __restrict__ pw,
                                                         __bf16* __restrict__ wbt) {
    __shared__ float tile[32][33];
    const int tx = threadIdx.x, ty = threadIdx.y;      // 32 x 8
    const int n0 = blockIdx.x * 32, k0 = blockIdx.y * 32;
#pragma unroll
    for (int i = 0; i < 4; ++i)
        tile[ty + i * 8][tx] = pw[(size_t)(k0 + ty + i * 8) * QKVN + n0 + tx];
    __syncthreads();
#pragma unroll
    for (int i = 0; i < 4; ++i)
        wbt[(size_t)(n0 + ty + i * 8) * HID + k0 + tx] = (__bf16)tile[tx][ty + i * 8];
}

// ================= kernel 0b: hash_w[d][j] -> hwT[j][d] bf16 =================
__global__ __launch_bounds__(256) void k_transpose_hashw(const float* __restrict__ hw,
                                                         __bf16* __restrict__ hwt) {
    const int idx = blockIdx.x * 256 + threadIdx.x;    // 16384 total
    const int j = idx >> 7, d = idx & 127;
    hwt[j * HD + d] = (__bf16)hw[d * RBIT + j];
}

// ================= kernel 1: qk = hidden @ proj_w[:, :5120] + bias (bf16 WMMA) ===
// Macro tile 128(M) x 256(N) per 256-thread block; 8 waves in 2(M) x 4(N) grid;
// each wave owns a 64x64 register tile (4x4 WMMA tiles) -> 16 wmma : 16 ds_load
// per K-step.
#define LSTR 40   // padded LDS stride (bf16 elems) -> conflict-free b128 frag loads
__global__ __launch_bounds__(256) void k_qkv_gemm(const float* __restrict__ hidden,
                                                  const __bf16* __restrict__ wbt,
                                                  const float* __restrict__ bias,
                                                  float* __restrict__ qk) {
    __shared__ __bf16 As[2][128 * LSTR];   // 10 KB each
    __shared__ __bf16 Bs[2][256 * LSTR];   // 20 KB each  (total 60 KB)
    const int tid  = threadIdx.x;
    const int lane = tid & 31, wave = tid >> 5;
    const int l16  = lane & 15, half = lane >> 4;
    const int wm = (wave & 1) * 64;        // 2 waves along M
    const int wn = (wave >> 1) * 64;       // 4 waves along N
    const int m0 = blockIdx.y * 128, n0 = blockIdx.x * 256;

    v8f zero = {};
    v8f acc[4][4];
#pragma unroll
    for (int i = 0; i < 4; ++i)
#pragma unroll
        for (int j = 0; j < 4; ++j) acc[i][j] = zero;

    auto stage = [&](int buf, int kt) {
        const int k0 = kt * 32;
        // A: 128x32 fp32 -> bf16 (4 chunks of 4 floats per thread)
#pragma unroll
        for (int i = 0; i < 4; ++i) {
            const int c = tid + i * 256;
            const int row = c >> 3, ko = (c & 7) * 4;
            float4 f = *(const float4*)(hidden + (size_t)(m0 + row) * HID + k0 + ko);
            v4bf o;
            o[0] = (__bf16)f.x; o[1] = (__bf16)f.y; o[2] = (__bf16)f.z; o[3] = (__bf16)f.w;
            *(v4bf*)(&As[buf][row * LSTR + ko]) = o;
        }
        // B: 256x32 bf16 copy (already transposed: row = N, contiguous K)
#pragma unroll
        for (int i = 0; i < 4; ++i) {
            const int c = tid + i * 256;
            const int row = c >> 2, ko = (c & 3) * 8;
            v8bf v = *(const v8bf*)(wbt + (size_t)(n0 + row) * HID + k0 + ko);
            *(v8bf*)(&Bs[buf][row * LSTR + ko]) = v;
        }
    };

    stage(0, 0);
    const int NK = HID / 32;   // 128
    for (int kt = 0; kt < NK; ++kt) {
        const int cur = kt & 1;
        __syncthreads();
        if (kt + 1 < NK) stage(cur ^ 1, kt + 1);
        v16bf a[4];
#pragma unroll
        for (int tm = 0; tm < 4; ++tm)
            a[tm] = load_frag_a(&As[cur][0], wm + tm * 16 + l16, half, LSTR);
#pragma unroll
        for (int tn = 0; tn < 4; ++tn) {
            v16bf b = load_frag_b(&Bs[cur][0], wn + tn * 16 + l16, half, LSTR);
#pragma unroll
            for (int tm = 0; tm < 4; ++tm)
                acc[tm][tn] = __builtin_amdgcn_wmma_f32_16x16x32_bf16(
                    false, a[tm], false, b, (short)0, acc[tm][tn], false, false);
        }
    }
    // C layout: acc[v] <-> (M = v + 8*half, N = l16)
#pragma unroll
    for (int tm = 0; tm < 4; ++tm)
#pragma unroll
        for (int tn = 0; tn < 4; ++tn) {
            const int col = n0 + wn + tn * 16 + l16;
            const float bv = bias[col];
#pragma unroll
            for (int v = 0; v < 8; ++v) {
                const int row = m0 + wm + tm * 16 + half * 8 + v;
                qk[(size_t)row * NQK + col] = acc[tm][tn][v] + bv;
            }
        }
}

// ================= kernel 2: attn dot/norms + WMMA hash + hamming ================
#define RSTR 136  // padded LDS stride
__global__ __launch_bounds__(64) void k_hash_epilogue(const float* __restrict__ qk,
                                                      const __bf16* __restrict__ hwt,
                                                      float* __restrict__ out) {
    __shared__ __bf16 Hs[128 * RSTR];       // hash_w^T  (j rows, K contiguous)
    __shared__ __bf16 Qs[2][16 * RSTR];     // per-wave 16 q rows (bf16)
    __shared__ __bf16 Ks[2][16 * RSTR];     // per-wave 16 k rows (bf16)
    __shared__ float  Nrm[2][16];

    const int tid  = threadIdx.x;           // 64 threads = 2 waves
    const int lane = tid & 31, w = tid >> 5;
    const int l16  = lane & 15, half = lane >> 4;
    const int pbase = (blockIdx.x * 2 + w) * 16;   // 16 pairs per wave
    const int b = pbase >> 5;                      // H=32 -> same b for all 16
    const int hbase = pbase & 31;                  // 0 or 16

    // stage hash weights into LDS (block-wide)
#pragma unroll
    for (int i = 0; i < 32; ++i) {
        const int c = tid + i * 64;                // 2048 chunks of 8 bf16
        const int row = c >> 4, ko = (c & 15) * 8;
        *(v8bf*)(&Hs[row * RSTR + ko]) = *(const v8bf*)(hwt + row * HD + ko);
    }

    // stage q/k rows + fused dot / norms (lane L: row r=L%16, cols half*64..+63)
    const int r = l16;
    const int h = hbase + r;
    const float* qrow = qk + (size_t)b * NQK + h * HD;
    const float* krow = qk + (size_t)b * NQK + NH * HD + (h >> 2) * HD;
    float dp = 0.f, qn = 0.f, kn = 0.f;
#pragma unroll
    for (int c = 0; c < 16; ++c) {
        const int col = half * 64 + c * 4;
        float4 q4 = *(const float4*)(qrow + col);
        float4 k4 = *(const float4*)(krow + col);
        dp += q4.x * k4.x + q4.y * k4.y + q4.z * k4.z + q4.w * k4.w;
        qn += q4.x * q4.x + q4.y * q4.y + q4.z * q4.z + q4.w * q4.w;
        kn += k4.x * k4.x + k4.y * k4.y + k4.z * k4.z + k4.w * k4.w;
        v4bf qb, kb;
        qb[0] = (__bf16)q4.x; qb[1] = (__bf16)q4.y; qb[2] = (__bf16)q4.z; qb[3] = (__bf16)q4.w;
        kb[0] = (__bf16)k4.x; kb[1] = (__bf16)k4.y; kb[2] = (__bf16)k4.z; kb[3] = (__bf16)k4.w;
        *(v4bf*)(&Qs[w][r * RSTR + col]) = qb;
        *(v4bf*)(&Ks[w][r * RSTR + col]) = kb;
    }
    dp += __shfl_xor(dp, 16, 32);
    qn += __shfl_xor(qn, 16, 32);
    kn += __shfl_xor(kn, 16, 32);
    if (lane < 16) {
        out[pbase + r] = dp * INV_SQRT_D;          // attn_weights
        Nrm[w][r] = sqrtf(qn) * sqrtf(kn);
    }
    __syncthreads();

    // q@hwT and k@hwT : M=16 pairs, N=128 bits, K=128, bf16 WMMA
    v8f zero = {};
    v8f acc1[8], acc2[8];
#pragma unroll
    for (int nt = 0; nt < 8; ++nt) { acc1[nt] = zero; acc2[nt] = zero; }
#pragma unroll
    for (int kt = 0; kt < 4; ++kt) {
        const int k0 = kt * 32;
        v16bf aq = load_frag_a(&Qs[w][k0], l16, half, RSTR);
        v16bf ak = load_frag_a(&Ks[w][k0], l16, half, RSTR);
#pragma unroll
        for (int nt = 0; nt < 8; ++nt) {
            v16bf bf = load_frag_b(&Hs[k0], nt * 16 + l16, half, RSTR);
            acc1[nt] = __builtin_amdgcn_wmma_f32_16x16x32_bf16(
                false, aq, false, bf, (short)0, acc1[nt], false, false);
            acc2[nt] = __builtin_amdgcn_wmma_f32_16x16x32_bf16(
                false, ak, false, bf, (short)0, acc2[nt], false, false);
        }
    }

    // |(tanh+1)/2 - (tanh+1)/2| = 0.5*|tanh-tanh|  ->  ham_w = 1 - S/128
    float s[8] = {0.f, 0.f, 0.f, 0.f, 0.f, 0.f, 0.f, 0.f};
#pragma unroll
    for (int nt = 0; nt < 8; ++nt)
#pragma unroll
        for (int v = 0; v < 8; ++v)
            s[v] += fabsf(tanhf(acc1[nt][v]) - tanhf(acc2[nt][v]));
#pragma unroll
    for (int v = 0; v < 8; ++v) {
        float x = s[v];
        x += __shfl_xor(x, 1, 32);
        x += __shfl_xor(x, 2, 32);
        x += __shfl_xor(x, 4, 32);
        x += __shfl_xor(x, 8, 32);
        if (l16 == 0) {
            const int rr = v + (half << 3);
            out[NPAIR + pbase + rr] =
                (1.0f - x * (1.0f / (float)RBIT)) * Nrm[w][rr] * INV_SQRT_D;
        }
    }
}

// ================================ launcher ================================
extern "C" void kernel_launch(void* const* d_in, const int* in_sizes, int n_in,
                              void* d_out, int out_size, void* d_ws, size_t ws_size,
                              hipStream_t stream) {
    const float* hidden = (const float*)d_in[0];
    const float* proj_w = (const float*)d_in[1];
    const float* proj_b = (const float*)d_in[2];
    const float* hash_w = (const float*)d_in[3];
    float* out = (float*)d_out;

    char* ws = (char*)d_ws;
    __bf16* wbt = (__bf16*)ws;
    __bf16* hwt = (__bf16*)(ws + WBT_BYTES);
    float*  qkw = (float*)(ws + QK_OFF);

    // 0a: transpose+convert projection weights (first 5120 cols)
    k_transpose_projw<<<dim3(NQK / 32, HID / 32), dim3(32, 8), 0, stream>>>(proj_w, wbt);
    // 0b: transpose+convert hash weights
    k_transpose_hashw<<<dim3(64), dim3(256), 0, stream>>>(hash_w, hwt);
    // 1: main bf16 WMMA GEMM (128x256 macro tile, 64x64 per wave) -> qk fp32
    k_qkv_gemm<<<dim3(NQK / 256, BB / 128), dim3(256), 0, stream>>>(hidden, wbt, proj_b, qkw);
    // 2: attn dot/norms + WMMA hash epilogue -> outputs
    k_hash_epilogue<<<dim3(NPAIR / 32), dim3(64), 0, stream>>>(qkw, hwt, out);
}
